// TransformerBlock_58256936403643
// MI455X (gfx1250) — compile-verified, matrix operand order checked by reading, not run on hardware
//
// Performer transformer block for MI455X (gfx1250, wave32, WMMA + TDM).
// All large GEMMs run through v_wmma_f32_16x16x32_bf16 (f32 accum); bf16
// A-operands are staged into LDS by the Tensor Data Mover (tensor_load_to_lds
// with pad_enable reproducing the padded LDS pitch), f32 operands are staged
// with vectorized b128 loads + packed b64 LDS stores. Needs ~341 MB workspace.

#include <hip/hip_runtime.h>

#define DEVINL __device__ __forceinline__

typedef __bf16 bf16_t;
typedef bf16_t bf16x8  __attribute__((ext_vector_type(8)));
typedef bf16_t bf16x16 __attribute__((ext_vector_type(16)));
typedef float  f32x8   __attribute__((ext_vector_type(8)));
typedef float  f32x4   __attribute__((ext_vector_type(4)));
typedef unsigned short u16x4 __attribute__((ext_vector_type(4)));
typedef unsigned int   u32x2 __attribute__((ext_vector_type(2)));

// problem shape (fixed by the reference)
constexpr int Bb = 8, Ss = 4096, Ee = 512, Hh = 8, DhC = 64, Mm = 128, Ff = 2048;
constexpr int ROWS = Bb * Ss;        // 32768
constexpr int BHS  = Bb * Hh * Ss;   // 262144

// ---------- small helpers ----------
DEVINL unsigned short f2bfu(float f) {
  unsigned u = __builtin_bit_cast(unsigned, f);
  unsigned r = (u + 0x7FFFu + ((u >> 16) & 1u)) >> 16;  // round-to-nearest-even
  return (unsigned short)r;
}
DEVINL bf16_t f2bf(float f) { unsigned short s = f2bfu(f); return __builtin_bit_cast(bf16_t, s); }
DEVINL unsigned pack2bf(float a, float b) {
  return (unsigned)f2bfu(a) | ((unsigned)f2bfu(b) << 16);
}
DEVINL float bfu2f(unsigned short u) { unsigned x = ((unsigned)u) << 16; return __builtin_bit_cast(float, x); }
DEVINL unsigned f2ord(float f) {  // monotone float -> uint for atomicMax
  unsigned u = __builtin_bit_cast(unsigned, f);
  return (u & 0x80000000u) ? ~u : (u | 0x80000000u);
}
DEVINL float ord2f(unsigned u) {
  unsigned b = (u & 0x80000000u) ? (u & 0x7FFFFFFFu) : ~u;
  return __builtin_bit_cast(float, b);
}
DEVINL float lane_xor(float v, int mask) {  // wave32 butterfly via ds_bpermute_b32
  int i = __builtin_bit_cast(int, v);
  int j = __builtin_amdgcn_ds_bpermute((((int)(threadIdx.x & 31) ^ mask) << 2), i);
  return __builtin_bit_cast(float, j);
}
DEVINL float wave_max(float v) { for (int m = 16; m; m >>= 1) v = fmaxf(v, lane_xor(v, m)); return v; }
DEVINL float wave_sum(float v) { for (int m = 16; m; m >>= 1) v += lane_xor(v, m); return v; }

// ---------- TDM: async bf16 tile copy global -> LDS ----------
#if __has_builtin(__builtin_amdgcn_tensor_load_to_lds)
#define HAVE_TDM 1
typedef unsigned int tdm_u32x4 __attribute__((ext_vector_type(4)));
typedef int          tdm_i32x4 __attribute__((ext_vector_type(4)));
typedef int          tdm_i32x8 __attribute__((ext_vector_type(8)));
#endif

// ---------- generic WMMA GEMM ----------
constexpr int EPI_NONE = 0, EPI_QKV = 1, EPI_BIAS_RES = 2, EPI_RELU_BF16 = 3, EPI_DIV_O = 4;
constexpr int BM = 128, BN = 128, BK = 64, SK = BK + 8;  // SK pad keeps 16B alignment
// Precondition used to drop row-bounds predication: every launch has Mr % 128 == 0.

#ifdef HAVE_TDM
// 2D tile: BK bf16 per row (128 B = 32 DWORDs), BM rows, row stride lda elems.
// pad_enable: +4 DWORDs (16 B) after every 32 DWORDs => LDS pitch SK = BK + 8.
DEVINL void tdm_load_bf16_tile(const unsigned short* g, int ldaElems, void* ldsDst) {
  unsigned long long ga = (unsigned long long)g;
  unsigned lds = (unsigned)(unsigned long long)ldsDst;
  tdm_u32x4 g0;
  g0[0] = 1u;                                                  // count=1, no gather
  g0[1] = lds;                                                 // lds_addr
  g0[2] = (unsigned)ga;                                        // global_addr[31:0]
  g0[3] = (unsigned)((ga >> 32) & 0x01FFFFFFu) | (2u << 30);   // global_addr[56:32] | type=2
  tdm_i32x8 g1;
  g1[0] = (1 << 16) | (1 << 20) | (4 << 22) | (3 << 25);  // data_size=2B, pad_en, every 32dw, pad 4dw
  g1[1] = (BK & 0xFFFF) << 16;   // tensor_dim0 = BK (bits 63:48 -> low16 here)
  g1[2] = (BM & 0xFFFF) << 16;   // tensor_dim0 hi = 0 | tensor_dim1 = BM (low16)
  g1[3] = (BK << 16);            // tensor_dim1 hi = 0 | tile_dim0 = BK
  g1[4] = BM;                    // tile_dim1 = BM, tile_dim2 = 0
  g1[5] = ldaElems;              // tensor_dim0_stride[31:0] (data_size units)
  g1[6] = 0;                     // stride0 hi | stride1 lo
  g1[7] = 0;
  tdm_i32x4 gz = {0, 0, 0, 0};
#if defined(__clang_major__) && (__clang_major__ >= 23)
  tdm_i32x8 gz8 = {0, 0, 0, 0, 0, 0, 0, 0};
  __builtin_amdgcn_tensor_load_to_lds(g0, g1, gz, gz, gz8, 0);
#else
  __builtin_amdgcn_tensor_load_to_lds(g0, g1, gz, gz, 0);
#endif
}
#endif

template<int EPI, bool ABF16, bool TRANSA>
__global__ __launch_bounds__(256)
void wmma_gemm_kernel(const void* __restrict__ Ap, long strideA, int lda,
                      const float* __restrict__ Bp, long strideB, int ldb,
                      const float* __restrict__ bias,
                      const float* __restrict__ extra, long strideE,
                      void* __restrict__ Out, long strideO,
                      int Mr, int N, int K, float scale) {
  (void)Mr;
  __shared__ bf16_t As[BM * SK];
  __shared__ bf16_t Bs[BN * SK];
  const int tid = threadIdx.x;
  const int z = blockIdx.z;
  const int rowBase = blockIdx.y * BM;
  const int colBase = blockIdx.x * BN;
  const float* Bw = Bp + (size_t)z * strideB;
  const float* Af32 = nullptr;
  const unsigned short* Abf = nullptr;
  if constexpr (ABF16) Abf = (const unsigned short*)Ap + (size_t)z * strideA;
  else                 Af32 = (const float*)Ap + (size_t)z * strideA;

  const int wave = tid >> 5, lane = tid & 31;
  const int lr = lane & 15, half = lane >> 4;
  const int rowW = (wave >> 1) * 32;   // 4 waves along M
  const int colW = (wave & 1) * 64;    // 2 waves along N

  f32x8 acc[2][4] = {};

  for (int k0 = 0; k0 < K; k0 += BK) {
    __syncthreads();
    if (k0 + BK < K)  // global_prefetch_b8 for next B K-slab
      __builtin_prefetch(Bw + (size_t)(k0 + BK) * ldb + colBase + (tid & 127), 0, 1);

    // ---- stage A tile (BM x BK) as bf16, no predication ----
    if constexpr (ABF16 && !TRANSA) {
#ifdef HAVE_TDM
      if (wave == 0)  // Tensor Data Mover: DMA the strided bf16 tile into LDS
        tdm_load_bf16_tile(Abf + (size_t)rowBase * lda + k0, lda, (void*)As);
#else
#pragma unroll
      for (int i = 0; i < 8; ++i) {
        int idx = i * 256 + tid;
        int r = idx >> 4, c4 = (idx & 15) * 4;
        u16x4 vv = *(const u16x4*)(Abf + (size_t)(rowBase + r) * lda + k0 + c4);
        *(u16x4*)&As[r * SK + c4] = vv;
      }
#endif
    } else if constexpr (!TRANSA) {  // f32 A: b128 loads, packed b64 LDS stores
#pragma unroll
      for (int i = 0; i < 8; ++i) {
        int idx = i * 256 + tid;
        int r = idx >> 4, c4 = (idx & 15) * 4;
        f32x4 vv = *(const f32x4*)(Af32 + (size_t)(rowBase + r) * lda + k0 + c4);
        u32x2 pk; pk[0] = pack2bf(vv[0], vv[1]); pk[1] = pack2bf(vv[2], vv[3]);
        *(u32x2*)&As[r * SK + c4] = pk;
      }
    } else {  // TRANSA bf16 (A stored [K][Mr]): b64 loads along m, scatter rows
#pragma unroll
      for (int i = 0; i < 8; ++i) {
        int idx = i * 256 + tid;
        int r4 = (idx & 31) * 4, c = idx >> 5;
        u16x4 vv = *(const u16x4*)(Abf + (size_t)(k0 + c) * lda + rowBase + r4);
        As[(r4 + 0) * SK + c] = __builtin_bit_cast(bf16_t, (unsigned short)vv[0]);
        As[(r4 + 1) * SK + c] = __builtin_bit_cast(bf16_t, (unsigned short)vv[1]);
        As[(r4 + 2) * SK + c] = __builtin_bit_cast(bf16_t, (unsigned short)vv[2]);
        As[(r4 + 3) * SK + c] = __builtin_bit_cast(bf16_t, (unsigned short)vv[3]);
      }
    }

    // ---- stage B tile (BK x BN) transposed into [n][k]; clamp cols (N=64 case) ----
#pragma unroll
    for (int i = 0; i < 8; ++i) {
      int idx = i * 256 + tid;
      int n4 = (idx & 31) * 4, c = idx >> 5;
      int gn = colBase + n4;
      const float* src = Bw + (size_t)(k0 + c) * ldb + (gn < N ? gn : 0);
      f32x4 vv = *(const f32x4*)src;
      Bs[(n4 + 0) * SK + c] = f2bf(vv[0]);
      Bs[(n4 + 1) * SK + c] = f2bf(vv[1]);
      Bs[(n4 + 2) * SK + c] = f2bf(vv[2]);
      Bs[(n4 + 3) * SK + c] = f2bf(vv[3]);
    }
#ifdef HAVE_TDM
    if constexpr (ABF16 && !TRANSA) {
      if (wave == 0) __builtin_amdgcn_s_wait_tensorcnt(0);
    }
#endif
    __syncthreads();

    // ---- 2 WMMA K-steps per staged tile ----
#pragma unroll
    for (int kk = 0; kk < 2; ++kk) {
      const int kb = kk * 32;
      bf16x16 af[2], bfv[4];
#pragma unroll
      for (int tm = 0; tm < 2; ++tm) {
        const bf16_t* p = &As[(rowW + tm * 16 + lr) * SK + kb];
        bf16x8 lo = *(const bf16x8*)(p + half * 8);
        bf16x8 hi = *(const bf16x8*)(p + 16 + half * 8);
#pragma unroll
        for (int e = 0; e < 8; ++e) { af[tm][e] = lo[e]; af[tm][e + 8] = hi[e]; }
      }
#pragma unroll
      for (int tn = 0; tn < 4; ++tn) {
        const bf16_t* p = &Bs[(colW + tn * 16 + lr) * SK + kb];
        bf16x8 lo = *(const bf16x8*)(p + half * 8);
        bf16x8 hi = *(const bf16x8*)(p + 16 + half * 8);
#pragma unroll
        for (int e = 0; e < 8; ++e) { bfv[tn][e] = lo[e]; bfv[tn][e + 8] = hi[e]; }
      }
#pragma unroll
      for (int tm = 0; tm < 2; ++tm)
#pragma unroll
        for (int tn = 0; tn < 4; ++tn)
          acc[tm][tn] = __builtin_amdgcn_wmma_f32_16x16x32_bf16(
              false, af[tm], false, bfv[tn], (short)0, acc[tm][tn], false, false);
    }
  }

  // ---- epilogue: D layout m = r + half*8, n = lr ----
#pragma unroll
  for (int tm = 0; tm < 2; ++tm)
#pragma unroll
    for (int tn = 0; tn < 4; ++tn)
#pragma unroll
      for (int r = 0; r < 8; ++r) {
        int gr = rowBase + rowW + tm * 16 + half * 8 + r;
        int gc = colBase + colW + tn * 16 + lr;
        if (gc >= N) continue;  // rows always in range (Mr % 128 == 0)
        float v = acc[tm][tn][r];
        if constexpr (EPI == EPI_NONE) {
          ((float*)Out)[(size_t)z * strideO + (size_t)gr * N + gc] = v;
        } else if constexpr (EPI == EPI_QKV) {
          v = (v + bias[gc]) * scale;
          int b = gr >> 12, s = gr & (Ss - 1);
          int h = gc >> 6,  d = gc & (DhC - 1);
          ((float*)Out)[(((size_t)(b * Hh + h)) * Ss + s) * DhC + d] = v;
        } else if constexpr (EPI == EPI_BIAS_RES) {
          ((float*)Out)[(size_t)gr * N + gc] = v + bias[gc] + extra[(size_t)gr * N + gc];
        } else if constexpr (EPI == EPI_RELU_BF16) {
          ((unsigned short*)Out)[(size_t)gr * N + gc] = f2bfu(fmaxf(v + bias[gc], 0.f));
        } else {  // EPI_DIV_O: o[b, s, h*64+d] = num / den
          float dn = extra[(size_t)z * strideE + gr];
          int b = z >> 3, h = z & 7;
          ((float*)Out)[((size_t)b * Ss + gr) * Ee + h * DhC + gc] = v / dn;
        }
      }
}

// ---------- FAVOR+ feature kernels (one wave per row, omega staged in LDS) ----------
DEVINL void proj_row(const float* __restrict__ u, const float* __restrict__ Om, int lane,
                     float& a0, float& a1, float& a2, float& a3, float& sq) {
  a0 = a1 = a2 = a3 = sq = 0.f;
#pragma unroll 4
  for (int d = 0; d < DhC; ++d) {
    float ud = u[d];
    const float* om = &Om[d * Mm];
    a0 = fmaf(ud, om[lane],      a0);
    a1 = fmaf(ud, om[lane + 32], a1);
    a2 = fmaf(ud, om[lane + 64], a2);
    a3 = fmaf(ud, om[lane + 96], a3);
    sq = fmaf(ud, ud, sq);
  }
}

__global__ __launch_bounds__(256)
void phi_q_kernel(const float* __restrict__ q, const float* __restrict__ omega,
                  unsigned short* __restrict__ phiq) {
  __shared__ float Om[DhC * Mm];
  __shared__ float U[8 * DhC];
  const int tid = threadIdx.x;
  const long rowBase = (long)blockIdx.x * 8;
#pragma unroll
  for (int i = 0; i < 32; ++i) Om[i * 256 + tid] = omega[i * 256 + tid];
#pragma unroll
  for (int i = 0; i < 2; ++i)  U[i * 256 + tid] = q[rowBase * DhC + i * 256 + tid];
  __syncthreads();
  const int w = tid >> 5, lane = tid & 31;
  float a0, a1, a2, a3, sq;
  proj_row(&U[w * DhC], Om, lane, a0, a1, a2, a3, sq);
  float mx = wave_max(fmaxf(fmaxf(a0, a1), fmaxf(a2, a3)));  // diag cancels per-row
  const float inv = 0.088388347648318447f;  // M^-0.5
  unsigned short* o = phiq + (rowBase + w) * Mm;
  o[lane]      = f2bfu(__expf(a0 - mx) * inv);
  o[lane + 32] = f2bfu(__expf(a1 - mx) * inv);
  o[lane + 64] = f2bfu(__expf(a2 - mx) * inv);
  o[lane + 96] = f2bfu(__expf(a3 - mx) * inv);
}

__global__ __launch_bounds__(256)
void phik_max_kernel(const float* __restrict__ k, const float* __restrict__ omega,
                     unsigned int* __restrict__ kstab) {
  __shared__ float Om[DhC * Mm];
  __shared__ float U[8 * DhC];
  __shared__ float wred[8];
  const int tid = threadIdx.x;
  const long rowBase = (long)blockIdx.x * 8;
#pragma unroll
  for (int i = 0; i < 32; ++i) Om[i * 256 + tid] = omega[i * 256 + tid];
#pragma unroll
  for (int i = 0; i < 2; ++i)  U[i * 256 + tid] = k[rowBase * DhC + i * 256 + tid];
  __syncthreads();
  const int w = tid >> 5, lane = tid & 31;
  float a0, a1, a2, a3, sq;
  proj_row(&U[w * DhC], Om, lane, a0, a1, a2, a3, sq);
  float mx = wave_max(fmaxf(fmaxf(a0, a1), fmaxf(a2, a3)));
  if (lane == 0) wred[w] = mx - 0.5f * sq;
  __syncthreads();
  if (tid == 0) {
    float m = wred[0];
#pragma unroll
    for (int i = 1; i < 8; ++i) m = fmaxf(m, wred[i]);
    atomicMax(&kstab[(int)(rowBase >> 12)], f2ord(m));
  }
}

__global__ __launch_bounds__(256)
void phik_kernel(const float* __restrict__ k, const float* __restrict__ omega,
                 const unsigned int* __restrict__ kstab, unsigned short* __restrict__ phik) {
  __shared__ float Om[DhC * Mm];
  __shared__ float U[8 * DhC];
  const int tid = threadIdx.x;
  const long rowBase = (long)blockIdx.x * 8;
#pragma unroll
  for (int i = 0; i < 32; ++i) Om[i * 256 + tid] = omega[i * 256 + tid];
#pragma unroll
  for (int i = 0; i < 2; ++i)  U[i * 256 + tid] = k[rowBase * DhC + i * 256 + tid];
  __syncthreads();
  const int w = tid >> 5, lane = tid & 31;
  float a0, a1, a2, a3, sq;
  proj_row(&U[w * DhC], Om, lane, a0, a1, a2, a3, sq);
  float base = -0.5f * sq - ord2f(kstab[(int)(rowBase >> 12)]);
  const float inv = 0.088388347648318447f;
  unsigned short* o = phik + (rowBase + w) * Mm;
  o[lane]      = f2bfu(__expf(a0 + base) * inv);
  o[lane + 32] = f2bfu(__expf(a1 + base) * inv);
  o[lane + 64] = f2bfu(__expf(a2 + base) * inv);
  o[lane + 96] = f2bfu(__expf(a3 + base) * inv);
}

__global__ void init_kstab_kernel(unsigned int* __restrict__ kstab) {
  if (threadIdx.x < Bb * Hh) kstab[threadIdx.x] = f2ord(-3.0e38f);
}

__global__ __launch_bounds__(256)
void ksum_kernel(const unsigned short* __restrict__ phik, float* __restrict__ ksum) {
  const int bh = blockIdx.x, tid = threadIdx.x;
  const int m = tid & 127, hlf = tid >> 7;
  const unsigned short* p = phik + (size_t)bh * Ss * Mm;
  float s = 0.f;
  for (int r = hlf; r < Ss; r += 2) s += bfu2f(p[(size_t)r * Mm + m]);
  __shared__ float red[256];
  red[tid] = s;
  __syncthreads();
  if (tid < 128) ksum[bh * Mm + tid] = red[tid] + red[tid + 128];
}

__global__ __launch_bounds__(256)
void den_kernel(const unsigned short* __restrict__ phiq, const float* __restrict__ ksum,
                float* __restrict__ den) {
  const int tid = threadIdx.x;
  const long rowBase = (long)blockIdx.x * 8;
  const int bh = (int)(rowBase >> 12);
  __shared__ float Ks[Mm];
  if (tid < Mm) Ks[tid] = ksum[bh * Mm + tid];
  __syncthreads();
  const int w = tid >> 5, lane = tid & 31;
  const unsigned short* p = phiq + (rowBase + w) * Mm;
  float s = 0.f;
#pragma unroll
  for (int j = 0; j < 4; ++j) { int m = lane + 32 * j; s += bfu2f(p[m]) * Ks[m]; }
  s = wave_sum(s);
  if (lane == 0) den[rowBase + w] = s + 1e-6f;
}

// ---------- LayerNorm over E=512, one block per row ----------
__global__ __launch_bounds__(256)
void ln_kernel(const float* __restrict__ in, const float* __restrict__ g,
               const float* __restrict__ b, float* __restrict__ out) {
  const int tid = threadIdx.x;
  const size_t base = (size_t)blockIdx.x * Ee;
  float v0 = in[base + tid], v1 = in[base + tid + 256];
  __shared__ float red[256];
  red[tid] = v0 + v1;
  __syncthreads();
  for (int off = 128; off; off >>= 1) { if (tid < off) red[tid] += red[tid + off]; __syncthreads(); }
  float mu = red[0] * (1.f / Ee);
  __syncthreads();
  float d0 = v0 - mu, d1 = v1 - mu;
  red[tid] = d0 * d0 + d1 * d1;
  __syncthreads();
  for (int off = 128; off; off >>= 1) { if (tid < off) red[tid] += red[tid + off]; __syncthreads(); }
  float rs = rsqrtf(red[0] * (1.f / Ee) + 1e-6f);
  out[base + tid]       = d0 * rs * g[tid]       + b[tid];
  out[base + tid + 256] = d1 * rs * g[tid + 256] + b[tid + 256];
}

// ---------- launcher ----------
extern "C" void kernel_launch(void* const* d_in, const int* in_sizes, int n_in,
                              void* d_out, int out_size, void* d_ws, size_t ws_size,
                              hipStream_t stream) {
  (void)in_sizes; (void)n_in; (void)out_size; (void)ws_size;
  const float* x   = (const float*)d_in[0];
  const float* Wq  = (const float*)d_in[1];
  const float* bq  = (const float*)d_in[2];
  const float* Wk  = (const float*)d_in[3];
  const float* bk  = (const float*)d_in[4];
  const float* Wv  = (const float*)d_in[5];
  const float* bv  = (const float*)d_in[6];
  const float* Wo  = (const float*)d_in[7];
  const float* bo  = (const float*)d_in[8];
  const float* om  = (const float*)d_in[9];
  const float* W1  = (const float*)d_in[10];
  const float* b1  = (const float*)d_in[11];
  const float* W2  = (const float*)d_in[12];
  const float* b2  = (const float*)d_in[13];
  const float* g1  = (const float*)d_in[14];
  const float* be1 = (const float*)d_in[15];
  const float* g2  = (const float*)d_in[16];
  const float* be2 = (const float*)d_in[17];

  char* ws = (char*)d_ws;
  const size_t R64 = 64ull << 20;  // 64 MiB regions
  float* q            = (float*)(ws + 0 * R64);           // [B,H,S,Dh]
  float* kbuf         = (float*)(ws + 1 * R64);
  float* v            = (float*)(ws + 2 * R64);
  unsigned short* phq = (unsigned short*)(ws + 3 * R64);  // [B,H,S,M] bf16
  unsigned short* phk = (unsigned short*)(ws + 4 * R64);
  char* sm = ws + 5 * R64;
  float* kv          = (float*)sm;                        // [BH,128,64]
  float* ksum        = (float*)(sm + (2ull << 20));       // [BH,128]
  unsigned int* kst  = (unsigned int*)(sm + (2ull << 20) + (64ull << 10));
  float* den         = (float*)(sm + (4ull << 20));       // [BH,S]
  // dataflow aliases (regions dead before reuse)
  float* o    = q;                       // attn out pre-projection [B,S,512]
  float* xres = kbuf;                    // x + attn
  unsigned short* tff = phq;             // FFN hidden bf16, spans phq+phk (128 MiB)
  float* y    = v;                       // out1 + ffn pre-LN2
  float* out1 = (float*)d_out;           // staged in d_out, overwritten by final LN

  dim3 blk(256, 1, 1);
  const float sc = 0.35355339059327379f;  // Dh^-0.25

  dim3 gE(Ee / BN, ROWS / BM, 1);
  wmma_gemm_kernel<EPI_QKV, false, false><<<gE, blk, 0, stream>>>(
      x, 0, Ee, Wq, 0, Ee, bq, nullptr, 0, q, 0, ROWS, Ee, Ee, sc);
  wmma_gemm_kernel<EPI_QKV, false, false><<<gE, blk, 0, stream>>>(
      x, 0, Ee, Wk, 0, Ee, bk, nullptr, 0, kbuf, 0, ROWS, Ee, Ee, sc);
  wmma_gemm_kernel<EPI_QKV, false, false><<<gE, blk, 0, stream>>>(
      x, 0, Ee, Wv, 0, Ee, bv, nullptr, 0, v, 0, ROWS, Ee, Ee, 1.0f);

  init_kstab_kernel<<<1, 64, 0, stream>>>(kst);
  phi_q_kernel<<<BHS / 8, blk, 0, stream>>>(q, om, phq);
  phik_max_kernel<<<BHS / 8, blk, 0, stream>>>(kbuf, om, kst);
  phik_kernel<<<BHS / 8, blk, 0, stream>>>(kbuf, om, kst, phk);

  {  // kv[bh] = phi_k^T (128 x S) @ v (S x 64), batched over 64 heads
    dim3 g(1, 1, Bb * Hh);
    wmma_gemm_kernel<EPI_NONE, true, true><<<g, blk, 0, stream>>>(
        phk, (long)Ss * Mm, Mm, v, (long)Ss * DhC, DhC,
        nullptr, nullptr, 0, kv, (long)Mm * DhC, Mm, DhC, Ss, 1.f);
  }
  ksum_kernel<<<Bb * Hh, blk, 0, stream>>>(phk, ksum);
  den_kernel<<<BHS / 8, blk, 0, stream>>>(phq, ksum, den);
  {  // num[bh] = phi_q (S x 128) @ kv (128 x 64); epilogue divides by den -> o
    dim3 g(1, Ss / BM, Bb * Hh);
    wmma_gemm_kernel<EPI_DIV_O, true, false><<<g, blk, 0, stream>>>(
        phq, (long)Ss * Mm, Mm, kv, (long)Mm * DhC, DhC,
        nullptr, den, (long)Ss, o, 0, Ss, DhC, Mm, 1.f);
  }
  // output projection + residual with x
  wmma_gemm_kernel<EPI_BIAS_RES, false, false><<<gE, blk, 0, stream>>>(
      o, 0, Ee, Wo, 0, Ee, bo, x, 0, xres, 0, ROWS, Ee, Ee, 1.f);
  ln_kernel<<<ROWS, blk, 0, stream>>>(xres, g1, be1, out1);
  {  // FFN up + ReLU -> bf16 hidden
    dim3 g(Ff / BN, ROWS / BM, 1);
    wmma_gemm_kernel<EPI_RELU_BF16, false, false><<<g, blk, 0, stream>>>(
        out1, 0, Ee, W1, 0, Ff, b1, nullptr, 0, tff, 0, ROWS, Ff, Ee, 1.f);
  }
  // FFN down + residual with out1
  wmma_gemm_kernel<EPI_BIAS_RES, true, false><<<gE, blk, 0, stream>>>(
      tff, 0, Ff, W2, 0, Ee, b2, out1, 0, y, 0, ROWS, Ee, Ff, 1.f);
  ln_kernel<<<ROWS, blk, 0, stream>>>(y, g2, be2, (float*)d_out);
}
